// EfficientSelfAttention_12094627906077
// MI455X (gfx1250) — matrix operand order repaired and checked by reading, not compiled, not run on hardware
//
#include <hip/hip_runtime.h>
#include <hip/hip_bf16.h>

// ---------------------------------------------------------------------------
// EfficientSelfAttention (PVT SR-attention) for MI455X / gfx1250, bf16 WMMA.
// B=8, H=W=128, N=16384, C=128, HEADS=2, hd=64, SR=8, M=256.
// ---------------------------------------------------------------------------

typedef __attribute__((ext_vector_type(16))) __bf16 v16bf;
typedef __attribute__((ext_vector_type(8)))  __bf16 v8bf;
typedef __attribute__((ext_vector_type(4)))  __bf16 v4bf;
typedef __attribute__((ext_vector_type(8)))  float  v8f;

union AB { v16bf v; v8bf h[2]; };

static __device__ __forceinline__ __bf16 f2bf(float f) {
    union { float f; unsigned u; } in; in.f = f;
    unsigned u = in.u;
    u += 0x7FFFu + ((u >> 16) & 1u);            // round-to-nearest-even
    union { unsigned short s; __bf16 b; } out;
    out.s = (unsigned short)(u >> 16);
    return out.b;
}

static __device__ __forceinline__ v8f wmma_bf16(v16bf a, v16bf b, v8f c) {
    return __builtin_amdgcn_wmma_f32_16x16x32_bf16(
        /*neg_a=*/false, a, /*neg_b=*/false, b,
        /*c_mod=*/(short)0, c, /*reuse_a=*/false, /*reuse_b=*/false);
}

// ---------------------------------------------------------------------------
// 1) fp32 -> bf16 conversion (4 elements / thread)
// ---------------------------------------------------------------------------
__global__ __launch_bounds__(256) void cvt_f32_bf16(const float* __restrict__ s,
                                                    __bf16* __restrict__ d, int n) {
    int i = (blockIdx.x * 256 + threadIdx.x) * 4;
    if (i < n) {
        float4 f = *reinterpret_cast<const float4*>(s + i);
        v4bf o;
        o[0] = f2bf(f.x); o[1] = f2bf(f.y); o[2] = f2bf(f.z); o[3] = f2bf(f.w);
        *reinterpret_cast<v4bf*>(d + i) = o;
    }
}

// ---------------------------------------------------------------------------
// 2) Wsr [O=128][C=128][8][8] fp32 -> Wsr_t [O][pix*128 + c] bf16
// ---------------------------------------------------------------------------
__global__ __launch_bounds__(256) void wsr_transpose(const float* __restrict__ w,
                                                     __bf16* __restrict__ d) {
    int t = blockIdx.x * 256 + threadIdx.x;       // < 128*8192
    int o = t >> 13;
    int rem = t & 8191;
    int pix = rem >> 7;
    int c = rem & 127;
    d[t] = f2bf(w[(o * 128 + c) * 64 + pix]);
}

// ---------------------------------------------------------------------------
// 3) SR conv (as GEMM 2048x128 x K=8192) + bias + LayerNorm -> xr bf16 [2048][128]
//    One block = 16 patches x 128 out-channels; wave w owns o-tile w.
// ---------------------------------------------------------------------------
__global__ __launch_bounds__(256) void conv_ln_kernel(
    const __bf16* __restrict__ xb, const __bf16* __restrict__ wsr,
    const float* __restrict__ bsr, const float* __restrict__ gamma,
    const float* __restrict__ beta, __bf16* __restrict__ xr) {
    __shared__ float tile[16][128];
    const int lane = threadIdx.x & 31, wave = threadIdx.x >> 5;
    const int l15 = lane & 15, hi = lane >> 4;
    const int wg = blockIdx.x;                    // 0..127
    const int p = wg * 16 + l15;                  // patch (A-row) for this lane
    const int b = p >> 8, m = p & 255;
    const int mi = m >> 4, mj = m & 15;
    const int A0 = (b * 16384 + mi * 1024 + mj * 8) * 128;
    const int o = wave * 16 + l15;                // out-channel (B-col)
    const __bf16* brow = wsr + o * 8192;
    const int kb8 = hi * 8, kb16 = hi * 16;

    v8f acc = {};
    for (int kt = 0; kt < 256; ++kt) {
        AB a, bf;
        int K0 = kt * 32 + kb16;
        int K1 = K0 + 8;
        int p0 = K0 >> 7, c0 = K0 & 127;
        int p1 = K1 >> 7, c1 = K1 & 127;
        a.h[0] = *reinterpret_cast<const v8bf*>(xb + A0 + (p0 >> 3) * 16384 + (p0 & 7) * 128 + c0);
        a.h[1] = *reinterpret_cast<const v8bf*>(xb + A0 + (p1 >> 3) * 16384 + (p1 & 7) * 128 + c1);
        bf.v   = *reinterpret_cast<const v16bf*>(brow + kt * 32 + kb16);
        if (kt + 1 < 256)
            __builtin_prefetch(brow + (kt + 1) * 32 + kb16, 0, 0);
        acc = wmma_bf16(a.v, bf.v, acc);
    }
    float bo = bsr[o];
    #pragma unroll
    for (int r = 0; r < 8; ++r) tile[r + hi * 8][o] = acc[r] + bo;
    __syncthreads();

    // LayerNorm across C=128 per patch row; 16 threads per row.
    int row = threadIdx.x >> 4, s = threadIdx.x & 15;
    float v[8], sum = 0.f, sq = 0.f;
    #pragma unroll
    for (int j = 0; j < 8; ++j) {
        v[j] = tile[row][s + 16 * j];
        sum += v[j]; sq += v[j] * v[j];
    }
    #pragma unroll
    for (int msk = 1; msk < 16; msk <<= 1) {
        sum += __shfl_xor(sum, msk, 32);
        sq  += __shfl_xor(sq,  msk, 32);
    }
    float mean = sum * (1.f / 128.f);
    float var  = sq * (1.f / 128.f) - mean * mean;
    float rstd = rsqrtf(var + 1e-5f);
    int rg = (wg * 16 + row) * 128;
    #pragma unroll
    for (int j = 0; j < 8; ++j) {
        int c = s + 16 * j;
        xr[rg + c] = f2bf((v[j] - mean) * rstd * gamma[c] + beta[c]);
    }
}

// ---------------------------------------------------------------------------
// 4) q projection: qb[row][o] = x @ Wq^T + bq, bf16 out. One wave = 16 rows x 128 cols.
// ---------------------------------------------------------------------------
__global__ __launch_bounds__(256) void qproj_kernel(
    const __bf16* __restrict__ xb, const __bf16* __restrict__ wq,
    const float* __restrict__ bq, __bf16* __restrict__ qb) {
    const int lane = threadIdx.x & 31, wave = threadIdx.x >> 5;
    const int l15 = lane & 15, hi = lane >> 4;
    const int wgRow = (blockIdx.x * 8 + wave) * 16;
    const __bf16* arow = xb + (wgRow + l15) * 128;
    AB a[4];
    #pragma unroll
    for (int kt = 0; kt < 4; ++kt) {
        int K0 = kt * 32 + hi * 16;
        a[kt].h[0] = *reinterpret_cast<const v8bf*>(arow + K0);
        a[kt].h[1] = *reinterpret_cast<const v8bf*>(arow + K0 + 8);
    }
    for (int ot = 0; ot < 8; ++ot) {
        int o = ot * 16 + l15;
        const __bf16* brow = wq + o * 128;
        v8f acc = {};
        #pragma unroll
        for (int kt = 0; kt < 4; ++kt) {
            AB bf; bf.v = *reinterpret_cast<const v16bf*>(brow + kt * 32 + hi * 16);
            acc = wmma_bf16(a[kt].v, bf.v, acc);
        }
        float bias = bq[o];
        int base = (wgRow + hi * 8) * 128 + o;
        #pragma unroll
        for (int r = 0; r < 8; ++r) qb[base + r * 128] = f2bf(acc[r] + bias);
    }
}

// ---------------------------------------------------------------------------
// 5) kv projection: xr @ Wkv^T + bkv -> k [b,h,m,d] bf16 and v^T [b,h,d,m] bf16
// ---------------------------------------------------------------------------
__global__ __launch_bounds__(256) void kv_kernel(
    const __bf16* __restrict__ xr, const __bf16* __restrict__ wkv,
    const float* __restrict__ bkv, __bf16* __restrict__ kb,
    __bf16* __restrict__ vt) {
    const int lane = threadIdx.x & 31, wave = threadIdx.x >> 5;
    const int l15 = lane & 15, hi = lane >> 4;
    const int wgRow = (blockIdx.x * 8 + wave) * 16;      // over 2048 rows
    const __bf16* arow = xr + (wgRow + l15) * 128;
    AB a[4];
    #pragma unroll
    for (int kt = 0; kt < 4; ++kt) {
        int K0 = kt * 32 + hi * 16;
        a[kt].h[0] = *reinterpret_cast<const v8bf*>(arow + K0);
        a[kt].h[1] = *reinterpret_cast<const v8bf*>(arow + K0 + 8);
    }
    for (int ot = 0; ot < 16; ++ot) {
        int o = ot * 16 + l15;                           // 0..255
        const __bf16* brow = wkv + o * 128;
        v8f acc = {};
        #pragma unroll
        for (int kt = 0; kt < 4; ++kt) {
            AB bf; bf.v = *reinterpret_cast<const v16bf*>(brow + kt * 32 + hi * 16);
            acc = wmma_bf16(a[kt].v, bf.v, acc);
        }
        float bias = bkv[o];
        #pragma unroll
        for (int r = 0; r < 8; ++r) {
            int rowg = wgRow + hi * 8 + r;
            int b = rowg >> 8, m = rowg & 255;
            float val = acc[r] + bias;
            if (o < 128) {
                int h = o >> 6, dd = o & 63;
                kb[((b * 2 + h) * 256 + m) * 64 + dd] = f2bf(val);
            } else {
                int o2 = o - 128;
                int h = o2 >> 6, dd = o2 & 63;
                vt[((b * 2 + h) * 64 + dd) * 256 + m] = f2bf(val);
            }
        }
    }
}

// ---------------------------------------------------------------------------
// 6) Attention: per wave one 16-query tile. S[16,256] in regs (16 WMMA tiles),
//    shuffle softmax, P -> LDS bf16, reload as A-frags, attn @ v (32 WMMAs).
// ---------------------------------------------------------------------------
__global__ __launch_bounds__(128) void attn_kernel(
    const __bf16* __restrict__ qb, const __bf16* __restrict__ kb,
    const __bf16* __restrict__ vt, __bf16* __restrict__ obuf) {
    __shared__ __align__(32) __bf16 Plds[4][16][256];    // 32 KB
    const int lane = threadIdx.x & 31, wave = threadIdx.x >> 5;
    const int l15 = lane & 15, hi = lane >> 4;
    const int gt = blockIdx.x * 4 + wave;                // (b*2+h)*1024 + ntile
    const int bh = gt >> 10, nt = gt & 1023;
    const int b = bh >> 1, h = bh & 1;
    const int n0 = nt * 16;

    // q A-fragments (K = d, 0..63)
    const __bf16* qrow = qb + (b * 16384 + n0 + l15) * 128 + h * 64;
    AB aq[2];
    #pragma unroll
    for (int kt = 0; kt < 2; ++kt) {
        int d0 = kt * 32 + hi * 16;
        aq[kt].h[0] = *reinterpret_cast<const v8bf*>(qrow + d0);
        aq[kt].h[1] = *reinterpret_cast<const v8bf*>(qrow + d0 + 8);
    }

    // S = q @ k^T : 16 m-tiles, K-loop of 2
    const __bf16* kbase = kb + bh * 256 * 64;
    v8f sacc[16];
    #pragma unroll
    for (int t = 0; t < 16; ++t) {
        v8f s = {};
        const __bf16* krow = kbase + (t * 16 + l15) * 64;
        #pragma unroll
        for (int kt = 0; kt < 2; ++kt) {
            AB bk; bk.v = *reinterpret_cast<const v16bf*>(krow + kt * 32 + hi * 16);
            s = wmma_bf16(aq[kt].v, bk.v, s);
        }
        sacc[t] = s;
    }

    // softmax over m (row-wise); lane holds rows r + hi*8, col l15 per tile
    const float scale = 0.125f;                          // hd^-0.5
    float inv[8];
    #pragma unroll
    for (int r = 0; r < 8; ++r) {
        float mx = sacc[0][r];
        #pragma unroll
        for (int t = 1; t < 16; ++t) mx = fmaxf(mx, sacc[t][r]);
        #pragma unroll
        for (int msk = 1; msk < 16; msk <<= 1) mx = fmaxf(mx, __shfl_xor(mx, msk, 32));
        float sm = 0.f;
        #pragma unroll
        for (int t = 0; t < 16; ++t) {
            float e = __expf((sacc[t][r] - mx) * scale);
            sacc[t][r] = e; sm += e;
        }
        #pragma unroll
        for (int msk = 1; msk < 16; msk <<= 1) sm += __shfl_xor(sm, msk, 32);
        inv[r] = 1.f / sm;
    }

    // P -> LDS (bf16, natural [n][m] layout)
    #pragma unroll
    for (int t = 0; t < 16; ++t)
        #pragma unroll
        for (int r = 0; r < 8; ++r)
            Plds[wave][r + hi * 8][t * 16 + l15] = f2bf(sacc[t][r]);
    __syncthreads();

    // reload P as A-fragments (row = l15, K = m)
    AB pf[8];
    const __bf16* prow = &Plds[wave][l15][0];
    #pragma unroll
    for (int mt = 0; mt < 8; ++mt) {
        int m0 = mt * 32 + hi * 16;
        pf[mt].h[0] = *reinterpret_cast<const v8bf*>(prow + m0);
        pf[mt].h[1] = *reinterpret_cast<const v8bf*>(prow + m0 + 8);
    }

    // out = P @ v : 4 d-tiles x K-loop of 8
    const __bf16* vbase = vt + bh * 64 * 256;
    for (int dt = 0; dt < 4; ++dt) {
        v8f oacc = {};
        #pragma unroll
        for (int mt = 0; mt < 8; ++mt) {
            const __bf16* vrow = vbase + (dt * 16 + l15) * 256 + mt * 32 + hi * 16;
            AB bv; bv.v = *reinterpret_cast<const v16bf*>(vrow);
            oacc = wmma_bf16(pf[mt].v, bv.v, oacc);
        }
        int base = (b * 16384 + n0 + hi * 8) * 128 + h * 64 + dt * 16 + l15;
        #pragma unroll
        for (int r = 0; r < 8; ++r)
            obuf[base + r * 128] = f2bf(oacc[r] * inv[r]);
    }
}

// ---------------------------------------------------------------------------
// 7) Output projection: y = obuf @ Wp^T + bp, fp32 out.
// ---------------------------------------------------------------------------
__global__ __launch_bounds__(256) void oproj_kernel(
    const __bf16* __restrict__ ob, const __bf16* __restrict__ wp,
    const float* __restrict__ bp, float* __restrict__ out) {
    const int lane = threadIdx.x & 31, wave = threadIdx.x >> 5;
    const int l15 = lane & 15, hi = lane >> 4;
    const int wgRow = (blockIdx.x * 8 + wave) * 16;
    const __bf16* arow = ob + (wgRow + l15) * 128;
    AB a[4];
    #pragma unroll
    for (int kt = 0; kt < 4; ++kt) {
        int K0 = kt * 32 + hi * 16;
        a[kt].h[0] = *reinterpret_cast<const v8bf*>(arow + K0);
        a[kt].h[1] = *reinterpret_cast<const v8bf*>(arow + K0 + 8);
    }
    for (int ot = 0; ot < 8; ++ot) {
        int o = ot * 16 + l15;
        const __bf16* brow = wp + o * 128;
        v8f acc = {};
        #pragma unroll
        for (int kt = 0; kt < 4; ++kt) {
            AB bf; bf.v = *reinterpret_cast<const v16bf*>(brow + kt * 32 + hi * 16);
            acc = wmma_bf16(a[kt].v, bf.v, acc);
        }
        float bias = bp[o];
        int base = (wgRow + hi * 8) * 128 + o;
        #pragma unroll
        for (int r = 0; r < 8; ++r) out[base + r * 128] = acc[r] + bias;
    }
}

// ---------------------------------------------------------------------------
// Launch
// ---------------------------------------------------------------------------
extern "C" void kernel_launch(void* const* d_in, const int* in_sizes, int n_in,
                              void* d_out, int out_size, void* d_ws, size_t ws_size,
                              hipStream_t stream) {
    const float* x     = (const float*)d_in[0];
    const float* Wq    = (const float*)d_in[3];
    const float* bq    = (const float*)d_in[4];
    const float* Wkv   = (const float*)d_in[5];
    const float* bkv   = (const float*)d_in[6];
    const float* Wp    = (const float*)d_in[7];
    const float* bp    = (const float*)d_in[8];
    const float* Wsr   = (const float*)d_in[9];
    const float* bsr   = (const float*)d_in[10];
    const float* gamma = (const float*)d_in[11];
    const float* beta  = (const float*)d_in[12];
    float* out = (float*)d_out;

    char* ws = (char*)d_ws;
    const size_t NX = 8ull * 16384 * 128;          // 16,777,216
    __bf16* xb    = (__bf16*)(ws);                           // 32 MB
    __bf16* qb    = (__bf16*)(ws + 33554432);                // 32 MB
    __bf16* obuf  = (__bf16*)(ws + 67108864);                // 32 MB
    __bf16* wsr_t = (__bf16*)(ws + 100663296);               // 2 MB
    __bf16* wq_b  = (__bf16*)(ws + 102760448);               // 32 KB
    __bf16* wkv_b = (__bf16*)(ws + 102793216);               // 64 KB
    __bf16* wp_b  = (__bf16*)(ws + 102858752);               // 32 KB
    __bf16* xr_b  = (__bf16*)(ws + 102891520);               // 512 KB
    __bf16* kbuf  = (__bf16*)(ws + 103415808);               // 512 KB
    __bf16* vtb   = (__bf16*)(ws + 103940096);               // 512 KB

    // conversions
    cvt_f32_bf16<<<(int)(NX / 1024), 256, 0, stream>>>(x, xb, (int)NX);
    cvt_f32_bf16<<<16, 256, 0, stream>>>(Wq, wq_b, 16384);
    cvt_f32_bf16<<<32, 256, 0, stream>>>(Wkv, wkv_b, 32768);
    cvt_f32_bf16<<<16, 256, 0, stream>>>(Wp, wp_b, 16384);
    wsr_transpose<<<4096, 256, 0, stream>>>(Wsr, wsr_t);

    // SR conv + LayerNorm
    conv_ln_kernel<<<128, 256, 0, stream>>>(xb, wsr_t, bsr, gamma, beta, xr_b);
    // q projection
    qproj_kernel<<<1024, 256, 0, stream>>>(xb, wq_b, bq, qb);
    // kv projection
    kv_kernel<<<16, 256, 0, stream>>>(xr_b, wkv_b, bkv, kbuf, vtb);
    // attention
    attn_kernel<<<4096, 128, 0, stream>>>(qb, kbuf, vtb, obuf);
    // output projection
    oproj_kernel<<<1024, 256, 0, stream>>>(obuf, wp_b, bp, out);
}
